// GCNLayer_9543417331984
// MI455X (gfx1250) — matrix operand order repaired and checked by reading, not compiled, not run on hardware
//
#include <hip/hip_runtime.h>

// ---------------------------------------------------------------------------
// Problem constants (from reference): B,S,H,E,L = 8,2048,768,4096,8
// ---------------------------------------------------------------------------
constexpr int Bn = 8;
constexpr int Sn = 2048;
constexpr int Hn = 768;
constexpr int En = 4096;
constexpr int Ln = 8;

constexpr int TM   = 32;              // edge rows per block (2 WMMA M-subtiles)
constexpr int HPAD = Hn + 8;          // LDS row stride in bf16 (1552B: 16B-mult, avoids 16-way bank conflicts)
constexpr int NOUT = Bn * Sn * Hn;    // 12,582,912 output floats

typedef __bf16 bf16;
typedef __attribute__((ext_vector_type(16))) __bf16 v16bf;
typedef __attribute__((ext_vector_type(8)))  __bf16 bf16x8;
typedef __attribute__((ext_vector_type(4)))  __bf16 bf16x4;
typedef __attribute__((ext_vector_type(8)))  float  v8f;

// ---------------------------------------------------------------------------
// Kernel 1: zero output accumulator + zero bucket counters
// ---------------------------------------------------------------------------
__global__ __launch_bounds__(256) void k_zero(float* __restrict__ out,
                                              int* __restrict__ counts) {
  size_t gid = (size_t)blockIdx.x * blockDim.x + threadIdx.x;
  size_t i = gid * 4;
  if (i + 3 < (size_t)NOUT) {
    float4 z = make_float4(0.f, 0.f, 0.f, 0.f);
    *(float4*)(out + i) = z;
  }
  if (gid < (size_t)(Bn * Ln)) counts[gid] = 0;
}

// ---------------------------------------------------------------------------
// Kernel 2: convert W [L,H,H] f32 -> bf16 (done once per launch; L2-resident)
// ---------------------------------------------------------------------------
__global__ __launch_bounds__(256) void k_wconv(const float* __restrict__ Wf,
                                               bf16* __restrict__ Wb) {
  size_t i = ((size_t)blockIdx.x * blockDim.x + threadIdx.x) * 4;
  if (i + 3 < (size_t)Ln * Hn * Hn) {
    float4 f = *(const float4*)(Wf + i);
    bf16x4 v;
    v[0] = (bf16)f.x; v[1] = (bf16)f.y; v[2] = (bf16)f.z; v[3] = (bf16)f.w;
    *(bf16x4*)(Wb + i) = v;
  }
}

// ---------------------------------------------------------------------------
// Kernel 3: bucket valid edges by (batch, label)
// ---------------------------------------------------------------------------
__global__ __launch_bounds__(256) void k_bucket(const int* __restrict__ edges,
                                                int* __restrict__ counts,
                                                int* __restrict__ buckets) {
  int gid = blockIdx.x * blockDim.x + threadIdx.x;
  if (gid >= Bn * En) return;
  int b = gid / En, e = gid % En;
  const int* ed = edges + (size_t)gid * 3;
  int s = ed[0], t = ed[1], l = ed[2];
  if (l >= 0 && l < Ln && s >= 0 && s < Sn && t >= 0 && t < Sn) {
    int bk = b * Ln + l;
    int pos = atomicAdd(&counts[bk], 1);
    buckets[(size_t)bk * En + pos] = e;
  }
}

// ---------------------------------------------------------------------------
// Kernel 4: grouped GEMM over buckets.
//   block = one tile of up to 32 edges sharing (batch,label); 256 thr = 8 waves
//   Y[32, 768] = Xgathered[32, 768] @ W[l]^T + bias, scatter-add to targets.
//   Each wave owns 6 N-tiles (wave + 8*j) x 2 M-subtiles; each B fragment
//   (W slice from global/L2) feeds TWO wmma ops -> halves B traffic per FLOP.
// ---------------------------------------------------------------------------
__global__ __launch_bounds__(256) void k_gemm(const float* __restrict__ node,
                                              const int*   __restrict__ edges,
                                              const bf16*  __restrict__ Wb,
                                              const float* __restrict__ bias,
                                              const int*   __restrict__ counts,
                                              const int*   __restrict__ buckets,
                                              float* __restrict__ out) {
  const int bucket = blockIdx.y;          // 0 .. B*L-1
  const int tile   = blockIdx.x;          // 0 .. E/TM-1
  const int cnt    = counts[bucket];
  if (tile * TM >= cnt) return;
  const int rows = min(TM, cnt - tile * TM);
  const int b    = bucket / Ln;
  const int lab  = bucket % Ln;

  __shared__ __align__(16) bf16 As[TM][HPAD];   // gathered src rows, bf16
  __shared__ int tgt_s[TM];

  const int t     = threadIdx.x;
  const int row16 = t >> 4;      // 16 threads cooperate per row
  const int w     = t & 15;

  // -------- stage: gather TM src rows, convert f32 -> bf16 into LDS --------
#pragma unroll
  for (int rr = 0; rr < TM / 16; ++rr) {
    const int row = row16 + 16 * rr;
    const float* srcrow = nullptr;
    if (row < rows) {
      int eidx = buckets[(size_t)bucket * En + tile * TM + row];
      const int* ed = edges + ((size_t)b * En + eidx) * 3;
      if (w == 0) tgt_s[row] = ed[1];
      srcrow = node + ((size_t)b * Sn + ed[0]) * Hn;
    } else if (w == 0) {
      tgt_s[row] = 0;
    }
#pragma unroll
    for (int i = 0; i < Hn / (16 * 4); ++i) {   // 12 iters: 16 thr x float4
      int c = w * 4 + i * 64;
      bf16x4 v = {};
      if (srcrow) {
        float4 f = *(const float4*)(srcrow + c);
        v[0] = (bf16)f.x; v[1] = (bf16)f.y; v[2] = (bf16)f.z; v[3] = (bf16)f.w;
      }
      *(bf16x4*)&As[row][c] = v;
    }
  }
  __syncthreads();

  // -------- compute: 8 waves x 6 N-tiles x 2 M-subtiles, K chunks of 32 ----
  const int lane = t & 31;
  const int wave = t >> 5;
  const int half = lane >> 4;    // 0: lanes 0-15, 1: lanes 16-31
  const int lm   = lane & 15;

  v8f acc0[6], acc1[6];
#pragma unroll
  for (int j = 0; j < 6; ++j) {
    acc0[j] = (v8f){0.f,0.f,0.f,0.f,0.f,0.f,0.f,0.f};
    acc1[j] = (v8f){0.f,0.f,0.f,0.f,0.f,0.f,0.f,0.f};
  }

  const bf16* Wl = Wb + (size_t)lab * Hn * Hn;

  for (int kc = 0; kc < Hn / 32; ++kc) {
    // A fragments (16x32, contiguous k order shared with B):
    //   elems 0-15 : K = kc*32 + 16*half + e       (contiguous 16, matches B)
    int k0 = kc * 32 + 16 * half;
    bf16x8 lo0 = *(const bf16x8*)&As[lm][k0];
    bf16x8 hi0 = *(const bf16x8*)&As[lm][k0 + 8];
    v16bf a0 = __builtin_shufflevector(lo0, hi0,
        0,1,2,3,4,5,6,7,8,9,10,11,12,13,14,15);
    bf16x8 lo1 = *(const bf16x8*)&As[lm + 16][k0];
    bf16x8 hi1 = *(const bf16x8*)&As[lm + 16][k0 + 8];
    v16bf a1 = __builtin_shufflevector(lo1, hi1,
        0,1,2,3,4,5,6,7,8,9,10,11,12,13,14,15);

    // B fragment: lane holds output row n, K = kc*32 + 16*half .. +15
    int kb = kc * 32 + 16 * half;
#pragma unroll
    for (int j = 0; j < 6; ++j) {
      int n = (wave + 8 * j) * 16 + lm;
      v16bf bm = *(const v16bf*)(Wl + (size_t)n * Hn + kb);
      acc0[j] = __builtin_amdgcn_wmma_f32_16x16x32_bf16(
          false, a0, false, bm, (short)0, acc0[j], false, false);
      acc1[j] = __builtin_amdgcn_wmma_f32_16x16x32_bf16(
          false, a1, false, bm, (short)0, acc1[j], false, false);
    }
  }

  // -------- epilogue: bias + scatter-add (C layout: VGPR r -> M=r+8*half) --
#pragma unroll
  for (int j = 0; j < 6; ++j) {
    int n = (wave + 8 * j) * 16 + lm;
    float bv = bias[lab * Hn + n];
#pragma unroll
    for (int r = 0; r < 8; ++r) {
      int m0 = r + 8 * half;
      if (m0 < rows) {
        atomicAdd(out + ((size_t)b * Sn + tgt_s[m0]) * Hn + n, acc0[j][r] + bv);
      }
      int m1 = m0 + 16;
      if (m1 < rows) {
        atomicAdd(out + ((size_t)b * Sn + tgt_s[m1]) * Hn + n, acc1[j][r] + bv);
      }
    }
  }
}

// ---------------------------------------------------------------------------
// Kernel 5: final ReLU
// ---------------------------------------------------------------------------
__global__ __launch_bounds__(256) void k_relu(float* __restrict__ out) {
  size_t i = ((size_t)blockIdx.x * blockDim.x + threadIdx.x) * 4;
  if (i + 3 < (size_t)NOUT) {
    float4 v = *(float4*)(out + i);
    v.x = fmaxf(v.x, 0.f); v.y = fmaxf(v.y, 0.f);
    v.z = fmaxf(v.z, 0.f); v.w = fmaxf(v.w, 0.f);
    *(float4*)(out + i) = v;
  }
}

// ---------------------------------------------------------------------------
// Launcher
// ---------------------------------------------------------------------------
extern "C" void kernel_launch(void* const* d_in, const int* in_sizes, int n_in,
                              void* d_out, int out_size, void* d_ws, size_t ws_size,
                              hipStream_t stream) {
  const float* node  = (const float*)d_in[0];   // [B,S,H] f32
  const int*   edges = (const int*)  d_in[1];   // [B,E,3] i32
  const float* Wf    = (const float*)d_in[2];   // [L,H,H] f32
  const float* bias  = (const float*)d_in[3];   // [L,H]   f32
  float* out = (float*)d_out;                   // [B,S,H] f32

  // workspace layout
  char* ws = (char*)d_ws;
  bf16* Wb      = (bf16*)ws;                                  // 9,437,184 B
  int*  counts  = (int*)(ws + (size_t)Ln * Hn * Hn * 2);      // 256 B
  int*  buckets = counts + 64;                                // 64*E*4 = 1 MB

  const int nblk_out = (NOUT / 4 + 255) / 256;                    // 12288
  const int nblk_w   = ((Ln * Hn * Hn) / 4 + 255) / 256;          // 4608

  k_zero  <<<nblk_out, 256, 0, stream>>>(out, counts);
  k_wconv <<<nblk_w,   256, 0, stream>>>(Wf, Wb);
  k_bucket<<<(Bn * En + 255) / 256, 256, 0, stream>>>(edges, counts, buckets);

  dim3 g(En / TM, Bn * Ln);   // 128 tiles x 64 buckets, early-exit on count
  k_gemm  <<<g, 256, 0, stream>>>(node, edges, Wb, bias, counts, buckets, out);

  k_relu  <<<nblk_out, 256, 0, stream>>>(out);
}
